// Sub_Graph_82325933130356
// MI455X (gfx1250) — matrix operand order repaired and checked by reading, not compiled, not run on hardware
//
#include <hip/hip_runtime.h>
#include <hip/hip_bf16.h>

typedef __attribute__((ext_vector_type(16))) _Float16 v16h;
typedef __attribute__((ext_vector_type(8)))  _Float16 v8h;
typedef __attribute__((ext_vector_type(8)))  float    v8f;

#define EPSF 1e-5f

__device__ __forceinline__ v16h cat16(v8h lo, v8h hi) {
    v16h r;
#pragma unroll
    for (int t = 0; t < 8; ++t) { r[t] = lo[t]; r[t + 8] = hi[t]; }
    return r;
}

// B fragment (32x16 f16) for v_wmma_f32_16x16x32_f16 from transposed weights WT[n][k]
// (row stride kdim halves). Lane n=lane&15 (col), kgroup=lane>>4:
// half j -> K = kk*32 + kgroup*16 + j  (16 contiguous halves = 32B).
__device__ __forceinline__ v16h load_bfrag(const _Float16* __restrict__ wt, int kdim,
                                           int ntile, int kk, int lane) {
    int n  = ntile * 16 + (lane & 15);
    int kg = lane >> 4;
    const _Float16* p = wt + n * kdim + kk * 32 + kg * 16;
    v8h lo = *reinterpret_cast<const v8h*>(p);
    v8h hi = *reinterpret_cast<const v8h*>(p + 8);
    return cat16(lo, hi);
}

// A fragment (16x32 f16) from LDS row-major [64][S] halves.
// Lane: row = mt*16 + (lane&15), khalf = lane>>4:
// half j -> K = kk*32 + khalf*16 + j  (16 contiguous halves)
template <int S>
__device__ __forceinline__ v16h load_afrag(const _Float16* x, int mt, int kk, int lane) {
    int row = mt * 16 + (lane & 15);
    int kh  = lane >> 4;
    const _Float16* p = x + row * S + kk * 32 + kh * 16;
    v8h lo = *reinterpret_cast<const v8h*>(p);
    v8h hi = *reinterpret_cast<const v8h*>(p + 8);
    return cat16(lo, hi);
}

__device__ __forceinline__ v8f wmma_f16(v16h a, v16h b, v8f c) {
    return __builtin_amdgcn_wmma_f32_16x16x32_f16(
        /*neg_a=*/false, a, /*neg_b=*/false, b,
        /*c_mod=*/(short)0, c, /*reuse_a=*/false, /*reuse_b=*/false);
}

// ---- Pre-kernel: transpose + f32->f16 convert weights into workspace ----
// W1T[i][n][k] = W1[i][k][n]  (3*128*128)   W2T[i][n][k] = W2[i][k][n]  (3*64*128)
__global__ void conv_weights(const float* __restrict__ W1, const float* __restrict__ W2,
                             _Float16* __restrict__ w1t, _Float16* __restrict__ w2t) {
    int idx = blockIdx.x * blockDim.x + threadIdx.x;
    if (idx < 3 * 128 * 128) {
        int i = idx >> 14;
        int rem = idx & 16383;
        int n = rem >> 7, k = rem & 127;
        w1t[idx] = (_Float16)W1[i * 16384 + k * 128 + n];
    }
    if (idx < 3 * 64 * 128) {
        int i = idx / 8192;
        int rem = idx & 8191;
        int n = rem >> 7, k = rem & 127;
        w2t[idx] = (_Float16)W2[i * 8192 + k * 64 + n];
    }
}

// ---- Fused 3-layer MLP, one workgroup (8 waves) per batch element ----
// LDS regions alias across barrier-separated lifetimes:
//   bufA: xbuf (GEMM1 input)  <->  hbuf16 (GEMM2 input)
//   bufB: hbuf32 (GEMM1 f32 accum) <-> x1buf (GEMM2 f32 out)
__global__ __launch_bounds__(256)
void mlp_fused(const float* __restrict__ hidden,
               const float* __restrict__ b1,
               const float* __restrict__ gamma,
               const float* __restrict__ beta,
               const float* __restrict__ b2,
               const _Float16* __restrict__ w1t,
               const _Float16* __restrict__ w2t,
               float* __restrict__ out) {
    __shared__ _Float16 bufA[64][136];   // 17408 B
    __shared__ float    bufB[64][132];   // 33792 B
    __shared__ float    cmpart[4][64];   // partial col-max
    __shared__ float    colmax[64];

    _Float16 (*xbuf)[136]   = bufA;
    _Float16 (*hbuf16)[136] = bufA;
    float    (*hbuf32)[132] = bufB;
    float    (*x1buf)[68]   = reinterpret_cast<float(*)[68]>(&bufB[0][0]);

    const int tid  = threadIdx.x;
    const int lane = tid & 31;
    const int wv   = tid >> 5;           // wave id 0..7
    const size_t b = blockIdx.x;

    // ---- layer 0 input: global f32 -> LDS f16 ----
    {
        const float4* xg4 = reinterpret_cast<const float4*>(hidden + b * (64 * 128));
#pragma unroll
        for (int j = 0; j < 8; ++j) {
            int e4 = j * 256 + tid;
            float4 v = xg4[e4];
            int e = e4 * 4;
            int r = e >> 7, c = e & 127;
            xbuf[r][c + 0] = (_Float16)v.x;
            xbuf[r][c + 1] = (_Float16)v.y;
            xbuf[r][c + 2] = (_Float16)v.z;
            xbuf[r][c + 3] = (_Float16)v.w;
        }
    }
    __syncthreads();

    for (int layer = 0; layer < 3; ++layer) {
        const _Float16* W1l = w1t + layer * 128 * 128;
        const _Float16* W2l = w2t + layer * 64 * 128;

        // ===== GEMM1: h = x @ W1 + b1 ; wave wv owns N-tile wv (16 cols, all 64 rows)
        {
            v16h Bf[4];
#pragma unroll
            for (int kk = 0; kk < 4; ++kk) Bf[kk] = load_bfrag(W1l, 128, wv, kk, lane);
            const float bias = b1[layer * 128 + wv * 16 + (lane & 15)];
            const int half = lane >> 4;
            v8f acc[4];
#pragma unroll
            for (int mt = 0; mt < 4; ++mt) acc[mt] = (v8f){};
#pragma unroll
            for (int kk = 0; kk < 4; ++kk) {
#pragma unroll
                for (int mt = 0; mt < 4; ++mt) {
                    v16h Af = load_afrag<136>(&xbuf[0][0], mt, kk, lane);
                    acc[mt] = wmma_f16(Af, Bf[kk], acc[mt]);
                }
            }
            __syncthreads();  // xbuf fully consumed; bufA may be overwritten as hbuf16
#pragma unroll
            for (int mt = 0; mt < 4; ++mt) {
#pragma unroll
                for (int r = 0; r < 8; ++r) {
                    int m = mt * 16 + half * 8 + r;
                    hbuf32[m][wv * 16 + (lane & 15)] = acc[mt][r] + bias;
                }
            }
        }
        __syncthreads();

        // ===== LayerNorm over H + ReLU -> f16 (into bufA) ; 4 lanes per row
        {
            int row = tid >> 2;
            int sub = tid & 3;
            const float* gl = gamma + layer * 128;
            const float* bl = beta + layer * 128;
            float s = 0.f, s2 = 0.f;
#pragma unroll
            for (int j = 0; j < 32; ++j) {
                float v = hbuf32[row][sub * 32 + j];
                s += v; s2 += v * v;
            }
            s  += __shfl_xor(s, 1);  s  += __shfl_xor(s, 2);
            s2 += __shfl_xor(s2, 1); s2 += __shfl_xor(s2, 2);
            float mean = s * (1.f / 128.f);
            float var  = s2 * (1.f / 128.f) - mean * mean;
            float rstd = rsqrtf(var + EPSF);
#pragma unroll
            for (int j = 0; j < 32; ++j) {
                int c = sub * 32 + j;
                float v = (hbuf32[row][c] - mean) * rstd * gl[c] + bl[c];
                v = fmaxf(v, 0.f);
                hbuf16[row][c] = (_Float16)v;
            }
        }
        __syncthreads();  // hbuf32 consumed; bufB may be overwritten as x1buf

        // ===== GEMM2: x1 = h @ W2 + b2 ; wave -> (ntile = wv&3, two M-tiles)
        {
            const int nt  = wv & 3;
            const int mt0 = (wv >> 2) * 2;
            v16h Bf[4];
#pragma unroll
            for (int kk = 0; kk < 4; ++kk) Bf[kk] = load_bfrag(W2l, 128, nt, kk, lane);
            const float bias = b2[layer * 64 + nt * 16 + (lane & 15)];
            const int half = lane >> 4;
#pragma unroll
            for (int mi = 0; mi < 2; ++mi) {
                int mt = mt0 + mi;
                v8f acc = {};
#pragma unroll
                for (int kk = 0; kk < 4; ++kk) {
                    v16h Af = load_afrag<136>(&hbuf16[0][0], mt, kk, lane);
                    acc = wmma_f16(Af, Bf[kk], acc);
                }
#pragma unroll
                for (int r = 0; r < 8; ++r) {
                    int m = mt * 16 + half * 8 + r;
                    x1buf[m][nt * 16 + (lane & 15)] = acc[r] + bias;
                }
            }
        }
        __syncthreads();

        // ===== column max over L (seq pooling), 4-way parallel per column
        {
            int col = tid & 63, q = tid >> 6;
            float m = -3.402823466e38f;
#pragma unroll
            for (int l = q * 16; l < q * 16 + 16; ++l) m = fmaxf(m, x1buf[l][col]);
            cmpart[q][col] = m;
        }
        __syncthreads();
        if (tid < 64) {
            colmax[tid] = fmaxf(fmaxf(cmpart[0][tid], cmpart[1][tid]),
                                fmaxf(cmpart[2][tid], cmpart[3][tid]));
        }
        __syncthreads();

        if (layer < 2) {
            // next x = concat(x1, broadcast(colmax)) -> f16 (bufA, dead after GEMM2)
#pragma unroll
            for (int j = 0; j < 32; ++j) {
                int e = tid * 32 + j;
                int r = e >> 7, c = e & 127;
                float v = (c < 64) ? x1buf[r][c] : colmax[c - 64];
                xbuf[r][c] = (_Float16)v;
            }
            __syncthreads();
        }
    }

    // final: max over L of concat(x1, bcast(colmax)) == [colmax, colmax]
    if (tid < 128) out[b * 128 + tid] = colmax[tid & 63];
}

extern "C" void kernel_launch(void* const* d_in, const int* in_sizes, int n_in,
                              void* d_out, int out_size, void* d_ws, size_t ws_size,
                              hipStream_t stream) {
    const float* hidden = (const float*)d_in[0];
    const float* W1     = (const float*)d_in[1];
    const float* b1     = (const float*)d_in[2];
    const float* gamma  = (const float*)d_in[3];
    const float* beta   = (const float*)d_in[4];
    const float* W2     = (const float*)d_in[5];
    const float* b2     = (const float*)d_in[6];
    // d_in[7] = lengths (== L), unused

    _Float16* w1t = (_Float16*)d_ws;            // 3*128*128 halves
    _Float16* w2t = w1t + 3 * 128 * 128;        // 3*64*128 halves

    conv_weights<<<192, 256, 0, stream>>>(W1, W2, w1t, w2t);
    mlp_fused<<<8192, 256, 0, stream>>>(hidden, b1, gamma, beta, b2, w1t, w2t,
                                        (float*)d_out);
}
